// LocalLoss_47725676593641
// MI455X (gfx1250) — matrix operand order repaired and checked by reading, not compiled, not run on hardware
//
#include <hip/hip_runtime.h>
#include <hip/hip_bf16.h>
#include <math.h>

typedef __attribute__((ext_vector_type(16))) _Float16 v16h;
typedef __attribute__((ext_vector_type(8)))  _Float16 v8h;
typedef __attribute__((ext_vector_type(8)))  float    v8f;

#define N_ROWS   65536
#define DIM      256
#define GRP      1024
#define NTILES   (GRP / 16)
#define INV_BETA 10.0f
#define KEXP     14.4269504088896340f   /* 10 * log2(e): fixed softmax shift of 10 */
#define LN2F     0.6931471805599453f
#define EPSN     1e-12f

// ---------------- zero-init (graph-safe, deterministic) ----------------
__global__ void k_zero(float* __restrict__ sums, float* __restrict__ counts,
                       float* __restrict__ out) {
    int i = blockIdx.x * blockDim.x + threadIdx.x;
    if (i < GRP * DIM) sums[i] = 0.0f;
    if (i < GRP)       counts[i] = 0.0f;
    if (i == 0)        out[0] = 0.0f;
}

// ---------------- segment sums + counts via f32 atomics ----------------
// One block per row (D == 256): idx[n] is a scalar (uniform) load.
__global__ void k_accum(const float* __restrict__ x, const int* __restrict__ idx,
                        float* __restrict__ sums, float* __restrict__ counts) {
    int i = blockIdx.x * blockDim.x + threadIdx.x;   // 0 .. N*D-1
    int n = i >> 8;
    int d = i & 255;
    int g = idx[n];
    atomicAdd(&sums[g * DIM + d], x[i]);
    if (d == 0) atomicAdd(&counts[g], 1.0f);
}

// ---------------- group means -> L2 normalize -> f16 -------------------
__global__ void k_vnorm(const float* __restrict__ sums,
                        const float* __restrict__ counts,
                        _Float16* __restrict__ vh) {
    __shared__ float red[256];
    int g = blockIdx.x, d = threadIdx.x;
    float c = fmaxf(counts[g], 1.0f);
    float m = sums[g * DIM + d] / c;
    red[d] = m * m;
    __syncthreads();
    for (int s = 128; s > 0; s >>= 1) {
        if (d < s) red[d] += red[d + s];
        __syncthreads();
    }
    float nrm = fmaxf(sqrtf(red[0]), EPSN);
    vh[g * DIM + d] = (_Float16)(m / nrm);
}

// ---------------- x rows -> L2 normalize -> f16 -------------------------
__global__ void k_xnorm(const float* __restrict__ x, _Float16* __restrict__ xh) {
    __shared__ float red[256];
    int n = blockIdx.x, d = threadIdx.x;
    float v = x[(size_t)n * DIM + d];
    red[d] = v * v;
    __syncthreads();
    for (int s = 128; s > 0; s >>= 1) {
        if (d < s) red[d] += red[d + s];
        __syncthreads();
    }
    float nrm = fmaxf(sqrtf(red[0]), EPSN);
    xh[(size_t)n * DIM + d] = (_Float16)(v / nrm);
}

// ---------------- WMMA GEMM + fused log-softmax + NLL -------------------
// One wave owns a 16-row tile of xn (A in 64 VGPRs); loops over 64 column
// tiles of v with double-buffered B so the 8 v_wmma per tile overlap the
// next tile's 16 global_load_b128. Softmax uses the analytic bound
// |logit| <= 1/beta = 10 as a fixed shift (no running max needed), and the
// raw v_exp_f32 builtin since exp2 args lie in [-28.85, 0] (always normal).
__global__ __launch_bounds__(256) void k_gemm_loss(
        const _Float16* __restrict__ xh, const _Float16* __restrict__ vh,
        const int* __restrict__ idx, float* __restrict__ out) {
    const int lane    = threadIdx.x & 31;
    const int wave    = threadIdx.x >> 5;
    const int rowBase = (blockIdx.x * 8 + wave) * 16;   // 16 rows per wave
    const int lm = lane & 15;        // M row (A) / N col (B,C,D) within tile
    const int hk = lane >> 4;        // K subgroup select

    union V16 { v16h v; v8h h[2]; };

    // A tile: 16x256 f16, contiguous k order (same as B):
    // lanes 0-15: K = kc*32 + {0..15}; lanes 16-31: +16.
    V16 a[8];
    const _Float16* arow = xh + (size_t)(rowBase + lm) * DIM;
    #pragma unroll
    for (int kc = 0; kc < 8; ++kc) {
        int kb = kc * 32 + hk * 16;
        a[kc].h[0] = *(const v8h*)(arow + kb);
        a[kc].h[1] = *(const v8h*)(arow + kb + 8);
    }

    // Per C-register row (r + 8*hk): which column tile holds its target,
    // pre-gated on this lane owning the target's column (lm == tgt&15).
    int mcol[8];
    #pragma unroll
    for (int r = 0; r < 8; ++r) {
        int t = idx[rowBase + r + 8 * hk];
        mcol[r] = ((t & 15) == lm) ? (t >> 4) : -1;
    }

    float ssum[8], tdot[8];
    #pragma unroll
    for (int r = 0; r < 8; ++r) { ssum[r] = 0.0f; tdot[r] = 0.0f; }

    // B tile: v^T, 32x16 f16. contiguous k order (same as A): lane holds one column,
    // lanes 0-15 K=0..15, lanes 16-31 K=16..31 -> contiguous v16h.
    auto loadB = [&](v16h* b, int gt) {
        const _Float16* brow = vh + (size_t)(gt * 16 + lm) * DIM + hk * 16;
        #pragma unroll
        for (int kc = 0; kc < 8; ++kc)
            b[kc] = *(const v16h*)(brow + kc * 32);
    };
    auto computeT = [&](const v16h* b, int gt) {
        v8f c = {};
        #pragma unroll
        for (int kc = 0; kc < 8; ++kc)
            c = __builtin_amdgcn_wmma_f32_16x16x32_f16(
                    false, a[kc].v, false, b[kc], (short)0, c, false, false);
        #pragma unroll
        for (int r = 0; r < 8; ++r) {
            float d = c[r];                          // cosine dot in [-1,1]
            if (mcol[r] == gt) tdot[r] += d;
            // exp(10*(d-1)) = 2^(KEXP*d - KEXP); arg in [-28.85, 0] -> raw v_exp_f32
            ssum[r] += __builtin_amdgcn_exp2f(fmaf(d, KEXP, -KEXP));
        }
    };

    v16h b0[8], b1[8];
    loadB(b0, 0);
    for (int gt = 0; gt < NTILES; gt += 2) {
        loadB(b1, gt + 1);               // prefetch while computing gt
        computeT(b0, gt);
        if (gt + 2 < NTILES) loadB(b0, gt + 2);
        computeT(b1, gt + 1);
    }

    // sum ssum / tdot across the 16 lanes of each half-wave
    #pragma unroll
    for (int m = 1; m <= 8; m <<= 1) {
        #pragma unroll
        for (int r = 0; r < 8; ++r) {
            ssum[r] += __shfl_xor(ssum[r], m, 32);
            tdot[r] += __shfl_xor(tdot[r], m, 32);
        }
    }

    if (lm == 0) {  // lanes 0 and 16 each finalize 8 rows
        float acc = 0.0f;
        #pragma unroll
        for (int r = 0; r < 8; ++r)   // -logp = (10 + ln ssum) - 10*tdot
            acc += (INV_BETA + __builtin_amdgcn_logf(ssum[r]) * LN2F)
                   - INV_BETA * tdot[r];
        atomicAdd(out, acc * (1.0f / (float)N_ROWS));
    }
}

extern "C" void kernel_launch(void* const* d_in, const int* in_sizes, int n_in,
                              void* d_out, int out_size, void* d_ws, size_t ws_size,
                              hipStream_t stream) {
    const float* x   = (const float*)d_in[0];
    const int*   idx = (const int*)d_in[1];
    float*       out = (float*)d_out;

    char* ws = (char*)d_ws;
    float*    sums   = (float*)(ws);                 //  4 MB  [G*D]
    float*    counts = (float*)(ws + 4194304);       //  4 KB  [G]
    _Float16* vh     = (_Float16*)(ws + 4198400);    // 512 KB [G*D] (32B aligned)
    _Float16* xh     = (_Float16*)(ws + 4722688);    //  32 MB [N*D] (32B aligned)

    k_zero <<<(GRP * DIM + 255) / 256, 256, 0, stream>>>(sums, counts, out);
    k_accum<<<(N_ROWS * DIM) / 256,    256, 0, stream>>>(x, idx, sums, counts);
    k_vnorm<<<GRP,                     256, 0, stream>>>(sums, counts, vh);
    k_xnorm<<<N_ROWS,                  256, 0, stream>>>(x, xh);
    k_gemm_loss<<<N_ROWS / 128,        256, 0, stream>>>(xh, vh, idx, out);
}
